// StructureModule_73718818668717
// MI455X (gfx1250) — matrix operand order repaired and checked by reading, not compiled, not run on hardware
//
#include <hip/hip_runtime.h>
#include <hip/hip_bf16.h>

// ---------------- problem constants ----------------
#define Bc    2
#define Nc    512
#define CSc   384
#define CZc   128
#define Hc    12
#define Cc    16
#define PQc   4
#define PVc   8
#define PKVc  (PQc + PVc)     // 12
#define FEATc 2112            // H*C + H*PV*4 + H*CZ

typedef __attribute__((ext_vector_type(16))) _Float16 v16h;
typedef __attribute__((ext_vector_type(8)))  float    v8f;

static __device__ __forceinline__ int lane_id() { return (int)(threadIdx.x & 31u); }

static __device__ __forceinline__ v8f wmma_f16(v16h a, v16h b, v8f c) {
  // D = A(16x32 f16) * B(32x16 f16) + C(16x16 f32)
  return __builtin_amdgcn_wmma_f32_16x16x32_f16(false, a, false, b, (short)0, c, false, false);
}

// ---- A fragment: 16x32 f16 from an f32 source. Layout (matches B):
// lanes 0-15: element s -> K=s ; lanes 16-31: K=16+s  (M = lane&15)
static __device__ __forceinline__ v16h a_frag_f32(const float* base, int ms, int ks,
                                                  int kmax, int mmax) {
  int l = lane_id();
  int m = l & 15; if (m >= mmax) m = mmax - 1;   // clamp (padded rows unused)
  int kb = (l < 16) ? 0 : 16;
  v16h a;
#pragma unroll
  for (int i = 0; i < 8; ++i) {
    int k = kb + 2 * i;
    float x0 = (k     < kmax) ? base[(size_t)m * ms + (size_t)k * ks]       : 0.0f;
    float x1 = (k + 1 < kmax) ? base[(size_t)m * ms + (size_t)(k + 1) * ks] : 0.0f;
    a[2 * i]     = (_Float16)x0;
    a[2 * i + 1] = (_Float16)x1;
  }
  return a;
}

// A fragment from a f16 source, full K=32.
static __device__ __forceinline__ v16h a_frag_f16(const _Float16* base, int ms, int mmax) {
  int l = lane_id();
  int m = l & 15; if (m >= mmax) m = mmax - 1;
  int kb = (l < 16) ? 0 : 16;
  v16h a;
#pragma unroll
  for (int i = 0; i < 8; ++i) {
    int k = kb + 2 * i;
    a[2 * i]     = base[(size_t)m * ms + k];
    a[2 * i + 1] = base[(size_t)m * ms + k + 1];
  }
  return a;
}

// ---- B fragment: 32x16 f16 from an f32 source. Layout (matches A): N = lane&15,
// lanes 0-15 hold K=0..15 across the 16 halves, lanes 16-31 hold K=16..31.
static __device__ __forceinline__ v16h b_frag_f32(const float* base, int ks, int ns,
                                                  int kmax, int nmax) {
  int l = lane_id();
  int n = l & 15;
  bool nok = (n < nmax); if (!nok) n = nmax - 1;
  int kb = (l < 16) ? 0 : 16;
  v16h b;
#pragma unroll
  for (int h = 0; h < 16; ++h) {
    int k = kb + h;
    float x = (nok && k < kmax) ? base[(size_t)k * ks + (size_t)n * ns] : 0.0f;
    b[h] = (_Float16)x;
  }
  return b;
}

// ---- C/D tile store: VGPR r -> M = (lane<16?0:8)+r, N = lane&15
static __device__ __forceinline__ void c_store(float* base, int ms, int ns, v8f c,
                                               int mmax, int nmax) {
  int l = lane_id();
  int n = l & 15;
  if (n >= nmax) return;
  int mb = (l < 16) ? 0 : 8;
#pragma unroll
  for (int r = 0; r < 8; ++r) {
    int m = mb + r;
    if (m < mmax) base[(size_t)m * ms + (size_t)n * ns] = c[r];
  }
}

// ---------------- TDM (Tensor Data Mover) ----------------
#define USE_TDM 1
#if USE_TDM
typedef __attribute__((ext_vector_type(4))) unsigned int tdm_u32x4;
typedef __attribute__((ext_vector_type(8))) int          tdm_i32x8;
typedef __attribute__((ext_vector_type(4))) int          tdm_i32x4;

// 2D tile load: rows x cols f32 elements, row stride in elements, into LDS.
static __device__ __forceinline__ void tdm_load_tile_f32(const float* gsrc, float* ldst,
                                                         unsigned rows, unsigned cols,
                                                         unsigned row_stride_elems) {
  unsigned long long ga = (unsigned long long)gsrc;
  tdm_u32x4 g0;
  g0[0] = 1u;                                            // count=1, user descriptor
  g0[1] = (unsigned)(unsigned long long)ldst;            // lds_addr (bits 63:32)
  g0[2] = (unsigned)(ga & 0xFFFFFFFFull);                // global_addr lo
  g0[3] = ((unsigned)((ga >> 32) & 0x01FFFFFFull)) | 0x80000000u; // addr hi | type=2
  tdm_i32x8 g1;
  g1[0] = (int)(2u << 16);                               // data_size=4B, wg_mask=0
  g1[1] = (int)((cols & 0xFFFFu) << 16);                 // tensor_dim0 lo16 (bits 63:48)
  g1[2] = (int)(((cols >> 16) & 0xFFFFu) | ((rows & 0xFFFFu) << 16)); // dim0 hi | dim1 lo
  g1[3] = (int)(((rows >> 16) & 0xFFFFu) | ((cols & 0xFFFFu) << 16)); // dim1 hi | tile_dim0
  g1[4] = (int)(rows & 0xFFFFu);                         // tile_dim1 | tile_dim2=0
  g1[5] = (int)row_stride_elems;                         // tensor_dim0_stride lo32
  g1[6] = 0;
  g1[7] = 0;
  tdm_i32x4 g2 = (tdm_i32x4)0;
  tdm_i32x4 g3 = (tdm_i32x4)0;
#if __has_include(<hip/amd_detail/amd_gfx1250_TDM.h>)
  tdm_i32x8 g4 = (tdm_i32x8)0;
  __builtin_amdgcn_tensor_load_to_lds(g0, g1, g2, g3, g4, 0);   // clang-23 6-arg form
#else
  __builtin_amdgcn_tensor_load_to_lds(g0, g1, g2, g3, 0);       // ROCm 7.2 5-arg form
#endif
}
#endif

// ---------------- K1/K7: generic GEMM  Out[M x Nn] = X[M x K] @ W[Nn x K]^T + bias ----
__global__ void k_gemm_xwT(const float* __restrict__ X, const float* __restrict__ W,
                           const float* __restrict__ bias, float* __restrict__ Out,
                           int K, int Nn) {
  int m0 = blockIdx.x * 16;
  int n0 = blockIdx.y * 16;
  v8f acc = {};
  for (int k0 = 0; k0 < K; k0 += 32) {
    __builtin_prefetch(X + (size_t)m0 * K + k0 + 64, 0, 1);
    v16h a = a_frag_f32(X + (size_t)m0 * K + k0, K, 1, 32, 16);
    v16h b = b_frag_f32(W + (size_t)n0 * K + k0, 1, K, 32, 16);
    acc = wmma_f16(a, b, acc);
  }
  int n = lane_id() & 15;
  float bv = bias ? bias[n0 + n] : 0.0f;
#pragma unroll
  for (int r = 0; r < 8; ++r) acc[r] += bv;
  c_store(Out + (size_t)m0 * Nn + n0, Nn, 1, acc, 16, 16);
}

// ---------------- K2: rigid transforms -> global points + squared norms --------------
__global__ void k_points(const float* __restrict__ qp, const float* __restrict__ kvp,
                         const float* __restrict__ rot, const float* __restrict__ trans,
                         float* __restrict__ qg, float* __restrict__ kg,
                         float* __restrict__ vg, float* __restrict__ qn,
                         float* __restrict__ kn) {
  int t = blockIdx.x * blockDim.x + threadIdx.x;
  if (t >= Bc * Nc * Hc) return;
  int h = t % Hc;
  int bn = t / Hc;
  const float* R = rot + (size_t)bn * 9;
  const float* T = trans + (size_t)bn * 3;
  float r00=R[0],r01=R[1],r02=R[2],r10=R[3],r11=R[4],r12=R[5],r20=R[6],r21=R[7],r22=R[8];
  float t0=T[0], t1=T[1], t2=T[2];

  const float* qrow = qp + (size_t)bn * (3 * Hc * PQc);
  float* qgp = qg + ((size_t)bn * Hc + h) * (PQc * 3);
  float qs = 0.0f;
#pragma unroll
  for (int p = 0; p < PQc; ++p) {
    float lx = qrow[h * PQc + p];
    float ly = qrow[Hc * PQc + h * PQc + p];
    float lz = qrow[2 * Hc * PQc + h * PQc + p];
    float gx = r00*lx + r01*ly + r02*lz + t0;
    float gy = r10*lx + r11*ly + r12*lz + t1;
    float gz = r20*lx + r21*ly + r22*lz + t2;
    qgp[p*3+0] = gx; qgp[p*3+1] = gy; qgp[p*3+2] = gz;
    qs += gx*gx + gy*gy + gz*gz;
  }
  qn[(size_t)bn * Hc + h] = qs;

  const float* krow = kvp + (size_t)bn * (3 * Hc * PKVc);
  float* kgp = kg + ((size_t)bn * Hc + h) * (PQc * 3);
  float* vgp = vg + ((size_t)bn * Hc + h) * (PVc * 3);
  float ks = 0.0f;
#pragma unroll
  for (int pp = 0; pp < PKVc; ++pp) {
    float lx = krow[h * PKVc + pp];
    float ly = krow[Hc * PKVc + h * PKVc + pp];
    float lz = krow[2 * Hc * PKVc + h * PKVc + pp];
    float gx = r00*lx + r01*ly + r02*lz + t0;
    float gy = r10*lx + r11*ly + r12*lz + t1;
    float gz = r20*lx + r21*ly + r22*lz + t2;
    if (pp < PQc) {
      kgp[pp*3+0] = gx; kgp[pp*3+1] = gy; kgp[pp*3+2] = gz;
      ks += gx*gx + gy*gy + gz*gz;
    } else {
      int p = pp - PQc;
      vgp[p*3+0] = gx; vgp[p*3+1] = gy; vgp[p*3+2] = gz;
    }
  }
  kn[(size_t)bn * Hc + h] = ks;
}

// ---------------- K3: bias[b,h,i,j] = pair[b,i,j,:] . wb[h,:] + bb[h] ----------------
__global__ void k_bias(const float* __restrict__ pair, const float* __restrict__ wb,
                       const float* __restrict__ bb, float* __restrict__ bias) {
  int j0 = blockIdx.x * 16;
  int i  = blockIdx.y;
  int b  = blockIdx.z;
  const float* prow = pair + ((size_t)(b * Nc + i) * Nc + j0) * CZc;  // 16 rows x 128
  v8f acc = {};
  for (int k0 = 0; k0 < CZc; k0 += 32) {
    __builtin_prefetch(prow + 16 * CZc + k0, 0, 1);
    v16h a = a_frag_f32(prow + k0, CZc, 1, 32, 16);          // rows = j
    v16h w = b_frag_f32(wb + k0, 1, CZc, 32, Hc);            // cols = h
    acc = wmma_f16(a, w, acc);
  }
  int n = lane_id() & 15;
  float bv = (n < Hc) ? bb[n] : 0.0f;
#pragma unroll
  for (int r = 0; r < 8; ++r) acc[r] += bv;
  // store (m=j, n=h) -> bias[b,h,i,j]
  float* base = bias + (size_t)b * Hc * Nc * Nc + (size_t)i * Nc + j0;
  c_store(base, 1, Nc * Nc, acc, 16, Hc);
}

// ---------------- K4: logits[b,h,i,j] -------------------------------------------------
__global__ void k_logits(const float* __restrict__ q, const float* __restrict__ kv,
                         const float* __restrict__ qg, const float* __restrict__ kg,
                         const float* __restrict__ qn, const float* __restrict__ kn,
                         const float* __restrict__ bias, const float* __restrict__ hw,
                         float* __restrict__ logits) {
  int it = blockIdx.x >> 5;
  int jt = blockIdx.x & 31;
  int h = blockIdx.y;
  int b = blockIdx.z;
  int i0 = it * 16, j0 = jt * 16;

  v8f aqk = {};
  {
    v16h a = a_frag_f32(q + ((size_t)(b * Nc + i0) * Hc + h) * Cc, Hc * Cc, 1, Cc, 16);
    v16h k = b_frag_f32(kv + (size_t)(b * Nc + j0) * (2 * Hc * Cc) + h * 2 * Cc,
                        1, 2 * Hc * Cc, Cc, 16);
    aqk = wmma_f16(a, k, aqk);
  }
  v8f apt = {};
  {
    v16h a = a_frag_f32(qg + ((size_t)(b * Nc + i0) * Hc + h) * 12, Hc * 12, 1, 12, 16);
    v16h k = b_frag_f32(kg + ((size_t)(b * Nc + j0) * Hc + h) * 12, 1, Hc * 12, 12, 16);
    apt = wmma_f16(a, k, apt);
  }
  float gamma = log1pf(expf(hw[h]));           // softplus
  const float sqk = 0.14433756729740643f;      // sqrt(1/48)
  const float sb  = 0.57735026918962576f;      // sqrt(1/3)
  const float wpt = 0.13608276348795434f;      // sqrt(1/54)
  float ptc = -0.5f * wpt * gamma;

  int l = lane_id();
  int n = l & 15;
  int mb = (l < 16) ? 0 : 8;
  float knj = kn[(size_t)(b * Nc + j0 + n) * Hc + h];
  const float* brow = bias + (size_t)b * Hc * Nc * Nc + (size_t)h * Nc * Nc;
  float* outp = logits + (size_t)b * Hc * Nc * Nc + (size_t)h * Nc * Nc;
#pragma unroll
  for (int r = 0; r < 8; ++r) {
    int m = mb + r;
    float qni = qn[(size_t)(b * Nc + i0 + m) * Hc + h];
    float d2 = qni + knj - 2.0f * apt[r];
    float bv = brow[(size_t)(i0 + m) * Nc + (j0 + n)];
    outp[(size_t)(i0 + m) * Nc + (j0 + n)] = aqk[r] * sqk + sb * bv + ptc * d2;
  }
}

// ---------------- K4b: row softmax -> f16 probs ---------------------------------------
__global__ void k_softmax(const float* __restrict__ logits, _Float16* __restrict__ probs) {
  size_t row = blockIdx.x;                  // b*H*N rows
  const float* src = logits + row * Nc;
  _Float16* dst = probs + row * Nc;
  int l = (int)threadIdx.x;                 // 32 lanes, 16 elems each
  float v[16];
  float mx = -1e30f;
#pragma unroll
  for (int t = 0; t < 16; ++t) { v[t] = src[t * 32 + l]; mx = fmaxf(mx, v[t]); }
#pragma unroll
  for (int o = 16; o > 0; o >>= 1) mx = fmaxf(mx, __shfl_xor(mx, o));
  float s = 0.0f;
#pragma unroll
  for (int t = 0; t < 16; ++t) { v[t] = expf(v[t] - mx); s += v[t]; }
#pragma unroll
  for (int o = 16; o > 0; o >>= 1) s += __shfl_xor(s, o);
  float inv = 1.0f / s;
#pragma unroll
  for (int t = 0; t < 16; ++t) dst[t * 32 + l] = (_Float16)(v[t] * inv);
}

// ---------------- K5a: out_scalar = attn@v ; rp_g = attn@vg ---------------------------
__global__ void k_attn_v(const _Float16* __restrict__ probs, const float* __restrict__ kv,
                         const float* __restrict__ vg, float* __restrict__ feats,
                         float* __restrict__ rpg) {
  int i0 = blockIdx.x * 16;
  int h = blockIdx.y;
  int b = blockIdx.z;
  v8f av = {}, ag0 = {}, ag1 = {};
  const _Float16* A = probs + ((size_t)b * Hc + h) * Nc * Nc + (size_t)i0 * Nc;
  for (int k0 = 0; k0 < Nc; k0 += 32) {
    v16h a = a_frag_f16(A + k0, Nc, 16);
    // v[b,j,h,c] lives at kv[(b*N+j)*384 + h*32 + 16 + c]
    v16h bv = b_frag_f32(kv + (size_t)(b * Nc + k0) * 384 + h * 32 + 16, 384, 1, 32, 16);
    av = wmma_f16(a, bv, av);
    const float* VG = vg + (size_t)(b * Nc + k0) * (Hc * 24) + h * 24;
    v16h bg0 = b_frag_f32(VG, Hc * 24, 1, 32, 16);
    v16h bg1 = b_frag_f32(VG + 16, Hc * 24, 1, 32, 8);
    ag0 = wmma_f16(a, bg0, ag0);
    ag1 = wmma_f16(a, bg1, ag1);
  }
  c_store(feats + (size_t)(b * Nc + i0) * FEATc + h * Cc, FEATc, 1, av, 16, 16);
  float* G = rpg + ((size_t)(b * Nc + i0) * Hc + h) * 24;
  c_store(G, Hc * 24, 1, ag0, 16, 16);
  c_store(G + 16, Hc * 24, 1, ag1, 16, 8);
}

// ---------------- K5b: out_pair[b,i,h,:] = attn[b,:,i,:] @ pair[b,i,:,:] (TDM + LDS) --
__global__ void k_attn_pair(const _Float16* __restrict__ probs, const float* __restrict__ pair,
                            float* __restrict__ feats) {
  __shared__ float lds_pair[32 * CZc];      // 16 KB tile: 32 j-rows x 128
  int i = blockIdx.x;
  int b = blockIdx.y;
  int w = (int)(threadIdx.x >> 5);          // 8 waves, one 16-col N-tile each
  int n0 = w * 16;
  int l = lane_id();
  int n = l & 15;
  int kb = (l < 16) ? 0 : 16;
  v8f acc = {};
  const _Float16* A0 = probs + (size_t)b * Hc * Nc * Nc + (size_t)i * Nc;  // rows = h
  const float* P0 = pair + ((size_t)(b * Nc + i) * Nc) * CZc;
  for (int j0 = 0; j0 < Nc; j0 += 32) {
    __syncthreads();                        // previous tile fully consumed
#if USE_TDM
    if (w == 0) {
      tdm_load_tile_f32(P0 + (size_t)j0 * CZc, lds_pair, 32, CZc, CZc);
      __builtin_amdgcn_s_wait_tensorcnt(0);
    }
#else
    for (int t = (int)threadIdx.x; t < 32 * CZc; t += (int)blockDim.x)
      lds_pair[t] = P0[(size_t)j0 * CZc + t];
#endif
    if (j0 + 32 < Nc) __builtin_prefetch(P0 + (size_t)(j0 + 32) * CZc, 0, 1);
    __syncthreads();
    v16h a = a_frag_f16(A0 + j0, Nc * Nc, Hc);   // A rows = heads (12, padded)
    v16h bf;
#pragma unroll
    for (int hh = 0; hh < 16; ++hh)
      bf[hh] = (_Float16)lds_pair[(kb + hh) * CZc + n0 + n];
    acc = wmma_f16(a, bf, acc);
  }
  // (m=h, n=d) -> feats col 576 + h*128 + n0 + n
  c_store(feats + (size_t)(b * Nc + i) * FEATc + 576 + n0, CZc, 1, acc, Hc, 16);
}

// ---------------- K6: back to local frame + norms -------------------------------------
__global__ void k_local(const float* __restrict__ rpg, const float* __restrict__ rot,
                        const float* __restrict__ trans, float* __restrict__ feats) {
  int t = blockIdx.x * blockDim.x + threadIdx.x;
  if (t >= Bc * Nc * Hc * PVc) return;
  int p = t % PVc;
  int rest = t / PVc;
  int h = rest % Hc;
  int bn = rest / Hc;
  const float* R = rot + (size_t)bn * 9;
  const float* T = trans + (size_t)bn * 3;
  const float* G = rpg + ((size_t)bn * Hc + h) * 24 + p * 3;
  float dx = G[0] - T[0], dy = G[1] - T[1], dz = G[2] - T[2];
  // rp_l[o] = sum_p R[p][o] * d[p]  (R^T d, einsum 'bipo')
  float lx = R[0]*dx + R[3]*dy + R[6]*dz;
  float ly = R[1]*dx + R[4]*dy + R[7]*dz;
  float lz = R[2]*dx + R[5]*dy + R[8]*dz;
  float nr = sqrtf(lx*lx + ly*ly + lz*lz + 1e-8f);
  float* Frow = feats + (size_t)bn * FEATc;
  int col = h * PVc + p;
  Frow[192 + col] = lx;
  Frow[288 + col] = ly;
  Frow[384 + col] = lz;
  Frow[480 + col] = nr;
}

// ---------------- host-side launcher --------------------------------------------------
extern "C" void kernel_launch(void* const* d_in, const int* in_sizes, int n_in,
                              void* d_out, int out_size, void* d_ws, size_t ws_size,
                              hipStream_t stream) {
  const float* single = (const float*)d_in[0];
  const float* pair   = (const float*)d_in[1];
  const float* rot    = (const float*)d_in[2];
  const float* trans  = (const float*)d_in[3];
  const float* wq     = (const float*)d_in[4];
  const float* bq     = (const float*)d_in[5];
  const float* wkv    = (const float*)d_in[6];
  const float* bkv    = (const float*)d_in[7];
  const float* wqp    = (const float*)d_in[8];
  const float* bqp    = (const float*)d_in[9];
  const float* wkvp   = (const float*)d_in[10];
  const float* bkvp   = (const float*)d_in[11];
  const float* wb     = (const float*)d_in[12];
  const float* bb     = (const float*)d_in[13];
  const float* wo     = (const float*)d_in[14];
  const float* bo     = (const float*)d_in[15];
  const float* hw     = (const float*)d_in[16];
  float* out = (float*)d_out;
  float* ws  = (float*)d_ws;

  const size_t MN = (size_t)Bc * Nc;            // 1024
  float* q_     = ws;                           // 1024*192
  float* kv_    = q_    + MN * 192;             // 1024*384
  float* qp_    = kv_   + MN * 384;             // 1024*144
  float* kvp_   = qp_   + MN * 144;             // 1024*432
  float* qg_    = kvp_  + MN * 432;             // 1024*144
  float* kg_    = qg_   + MN * Hc * 12;         // 1024*144
  float* vg_    = kg_   + MN * Hc * 12;         // 1024*288
  float* qn_    = vg_   + MN * Hc * 24;         // 1024*12
  float* kn_    = qn_   + MN * Hc;              // 1024*12
  float* bias_  = kn_   + MN * Hc;              // 2*12*512*512
  float* logit_ = bias_ + (size_t)Bc * Hc * Nc * Nc;
  float* rpg_   = logit_+ (size_t)Bc * Hc * Nc * Nc;  // 1024*288
  float* feats_ = rpg_  + MN * Hc * 24;         // 1024*2112
  _Float16* probs_ = (_Float16*)(feats_ + MN * FEATc);

  dim3 w32(32, 1, 1);
  // K1: input projections  (M=1024, K=384)
  k_gemm_xwT<<<dim3(64, 192 / 16), w32, 0, stream>>>(single, wq,   bq,   q_,   CSc, 192);
  k_gemm_xwT<<<dim3(64, 384 / 16), w32, 0, stream>>>(single, wkv,  bkv,  kv_,  CSc, 384);
  k_gemm_xwT<<<dim3(64, 144 / 16), w32, 0, stream>>>(single, wqp,  bqp,  qp_,  CSc, 144);
  k_gemm_xwT<<<dim3(64, 432 / 16), w32, 0, stream>>>(single, wkvp, bkvp, kvp_, CSc, 432);
  // K2: rigid transforms
  k_points<<<(Bc * Nc * Hc + 255) / 256, 256, 0, stream>>>(qp_, kvp_, rot, trans,
                                                           qg_, kg_, vg_, qn_, kn_);
  // K3: pair bias (first pass over pair)
  k_bias<<<dim3(Nc / 16, Nc, Bc), w32, 0, stream>>>(pair, wb, bb, bias_);
  // K4: logits + softmax
  k_logits<<<dim3((Nc / 16) * (Nc / 16), Hc, Bc), w32, 0, stream>>>(q_, kv_, qg_, kg_,
                                                                    qn_, kn_, bias_, hw, logit_);
  k_softmax<<<Bc * Hc * Nc, w32, 0, stream>>>(logit_, probs_);
  // K5: attention applications
  k_attn_v<<<dim3(Nc / 16, Hc, Bc), w32, 0, stream>>>(probs_, kv_, vg_, feats_, rpg_);
  k_attn_pair<<<dim3(Nc, Bc), dim3(256, 1, 1), 0, stream>>>(probs_, pair, feats_);
  // K6: local frame + norms
  k_local<<<(Bc * Nc * Hc * PVc + 255) / 256, 256, 0, stream>>>(rpg_, rot, trans, feats_);
  // K7: output projection (K = 2112)
  k_gemm_xwT<<<dim3(64, 384 / 16), w32, 0, stream>>>(feats_, wo, bo, out, FEATc, 384);
}
